// SwinTransformerBlock_83193516524231
// MI455X (gfx1250) — matrix operand order repaired and checked by reading, not compiled, not run on hardware
//
#include <hip/hip_runtime.h>
#include <hip/hip_bf16.h>

// ---------------------------------------------------------------------------
// Swin block (shifted-window attention + LoRA + MLP) for gfx1250 / MI455X.
// All big GEMMs run through v_wmma_f32_16x16x32_f16 (f16 in, f32 accum),
// double-buffered LDS staging, 8 WMMAs per wave per K-step.
// ---------------------------------------------------------------------------

typedef _Float16 h16;
typedef __attribute__((ext_vector_type(16))) _Float16 v16h;
typedef __attribute__((ext_vector_type(8)))  _Float16 v8h;
typedef __attribute__((ext_vector_type(8)))  float    v8f;

union HV { v16h v; v8h h[2]; };

#define C_DIM 512
#define NHEAD 16
#define HD    32
#define NTOK  49
#define MROWS 50176   // 64 * 784

// A-fragment (16x32 f16, M rows in lanes): lane<16 holds K chunks [0..7],[16..23],
// lane>=16 holds [8..15],[24..31] of its row.
__device__ __forceinline__ v16h frag_a(const h16* rowptr, int lane) {
    int kb = (lane < 16) ? 0 : 8;
    HV u;
    u.h[0] = *(const v8h*)(rowptr + kb);
    u.h[1] = *(const v8h*)(rowptr + kb + 16);
    return u.v;
}
// B-fragment (32x16 f16, N cols in lanes): lane<16 holds K chunks [0..7],[16..23], lane>=16 [8..15],[24..31]
// of its column; colptr points at a [N][K]-layout row (32 contiguous halfs).
__device__ __forceinline__ v16h frag_b(const h16* colptr, int lane) {
    int kb = (lane < 16) ? 0 : 8;
    HV u;
    u.h[0] = *(const v8h*)(colptr + kb);
    u.h[1] = *(const v8h*)(colptr + kb + 16);
    return u.v;
}

// window-order row m  ->  source/destination token index (shift by +3 with wrap)
__device__ __forceinline__ int src_token(int m) {
    int j  = m % 7;
    int i  = (m / 7) % 7;
    int ww = (m / 49) & 3;
    int wh = (m / 196) & 3;
    int b  = m / 784;
    int hh = (wh * 7 + i + 3) % 28;
    int wc = (ww * 7 + j + 3) % 28;
    return b * 784 + hh * 28 + wc;
}

// -------------------------- LoRA weight fusion -----------------------------
__global__ __launch_bounds__(256) void fuse_w_kernel(
    const float* __restrict__ W, const float* __restrict__ la,
    const float* __restrict__ lb, h16* __restrict__ out, int Nn, int Kk) {
    int idx = blockIdx.x * 256 + threadIdx.x;
    if (idx >= Nn * Kk) return;
    int n = idx / Kk, k = idx - n * Kk;
    float v = W[idx];
    #pragma unroll
    for (int r = 0; r < 16; r++) v += lb[n * 16 + r] * la[r * Kk + k];
    out[idx] = (h16)v;
}

// ------------------- LayerNorm (+optional shift/partition) -----------------
__global__ __launch_bounds__(256) void ln_kernel(
    const float* __restrict__ x, const float* __restrict__ s,
    const float* __restrict__ b, h16* __restrict__ out, int permute) {
    __shared__ float red[256];
    int m = blockIdx.x;
    int t = permute ? src_token(m) : m;
    const float* xr = x + (size_t)t * C_DIM;
    int tid = threadIdx.x;
    float v0 = xr[tid], v1 = xr[tid + 256];
    red[tid] = v0 + v1;
    __syncthreads();
    for (int st = 128; st > 0; st >>= 1) {
        if (tid < st) red[tid] += red[tid + st];
        __syncthreads();
    }
    float mean = red[0] * (1.f / 512.f);
    __syncthreads();
    float d0 = v0 - mean, d1 = v1 - mean;
    red[tid] = d0 * d0 + d1 * d1;
    __syncthreads();
    for (int st = 128; st > 0; st >>= 1) {
        if (tid < st) red[tid] += red[tid + st];
        __syncthreads();
    }
    float rstd = rsqrtf(red[0] * (1.f / 512.f) + 1e-5f);
    h16* orow = out + (size_t)m * C_DIM;
    orow[tid]       = (h16)(d0 * rstd * s[tid]       + b[tid]);
    orow[tid + 256] = (h16)(d1 * rstd * s[tid + 256] + b[tid + 256]);
}

// ------------------------------ WMMA GEMM ----------------------------------
// out[M,N] = A[M,K](f16,row) * Bw[N,K](f16) + bias, fused epilogues.
// mode 0: store f16            mode 1: exact GELU, store f16
// mode 2: outF[m,n] = res[m,n] + v          (final residual, fp32)
// mode 3: t=src_token(m); outF[t,n] = res[t,n] + v   (window reverse+shift)
#define BM 128
#define BN 128
#define BK 32
__global__ __launch_bounds__(256) void gemm_wmma_kernel(
    const h16* __restrict__ A, const h16* __restrict__ Bw,
    const float* __restrict__ bias, int M, int N, int K, int mode,
    h16* __restrict__ outH, float* __restrict__ outF,
    const float* __restrict__ res) {
    __shared__ __align__(16) h16 As[2][BM * BK];
    __shared__ __align__(16) h16 Bs[2][BN * BK];
    int tid  = threadIdx.x;
    int lane = tid & 31;
    int wv   = tid >> 5;
    int mw   = wv & 1;   // wave M index 0..1 (64 rows each)
    int nw   = wv >> 1;  // wave N index 0..3 (32 cols each)
    int bm   = blockIdx.x * BM;
    int bn   = blockIdx.y * BN;
    int lrow = lane & 15;

    v8f acc[4][2] = {};

    // staging coordinates: each thread moves 16 halfs of A and 16 of B per tile
    int sr  = tid >> 1;            // 0..127
    int sko = (tid & 1) * 16;      // 0 or 16
    const h16* agp = A  + (size_t)(bm + sr) * K + sko;
    const h16* bgp = Bw + (size_t)(bn + sr) * K + sko;
    int nT = K / BK;

    // preload tile 0
    {
        v8h a0 = *(const v8h*)(agp),     a1 = *(const v8h*)(agp + 8);
        v8h b0 = *(const v8h*)(bgp),     b1 = *(const v8h*)(bgp + 8);
        *(v8h*)(&As[0][sr * BK + sko])     = a0;
        *(v8h*)(&As[0][sr * BK + sko + 8]) = a1;
        *(v8h*)(&Bs[0][sr * BK + sko])     = b0;
        *(v8h*)(&Bs[0][sr * BK + sko + 8]) = b1;
    }
    __syncthreads();

    for (int t = 0; t < nT; t++) {
        int cur = t & 1, nxt = cur ^ 1;
        v8h a0, a1, b0, b1;
        bool more = (t + 1) < nT;
        if (more) {  // issue global loads for tile t+1 early (latency hidden by WMMAs)
            const h16* ag = agp + (size_t)(t + 1) * BK;
            const h16* bg = bgp + (size_t)(t + 1) * BK;
            a0 = *(const v8h*)(ag); a1 = *(const v8h*)(ag + 8);
            b0 = *(const v8h*)(bg); b1 = *(const v8h*)(bg + 8);
            if (t + 2 < nT)  // global_prefetch_b8 for the tile after next
                __builtin_prefetch(agp + (size_t)(t + 2) * BK, 0, 0);
        }

        v16h af[4], bf[2];
        #pragma unroll
        for (int i = 0; i < 4; i++)
            af[i] = frag_a(&As[cur][(mw * 64 + i * 16 + lrow) * BK], lane);
        #pragma unroll
        for (int j = 0; j < 2; j++)
            bf[j] = frag_b(&Bs[cur][(nw * 32 + j * 16 + lrow) * BK], lane);
        #pragma unroll
        for (int i = 0; i < 4; i++)
            #pragma unroll
            for (int j = 0; j < 2; j++)
                acc[i][j] = __builtin_amdgcn_wmma_f32_16x16x32_f16(
                    false, af[i], false, bf[j], (short)0, acc[i][j], false, false);

        if (more) {
            *(v8h*)(&As[nxt][sr * BK + sko])     = a0;
            *(v8h*)(&As[nxt][sr * BK + sko + 8]) = a1;
            *(v8h*)(&Bs[nxt][sr * BK + sko])     = b0;
            *(v8h*)(&Bs[nxt][sr * BK + sko + 8]) = b1;
        }
        __syncthreads();
    }

    int rofs = (lane >= 16) ? 8 : 0;
    #pragma unroll
    for (int j = 0; j < 2; j++) {
        int col = bn + nw * 32 + j * 16 + lrow;
        float bb = bias ? bias[col] : 0.f;
        #pragma unroll
        for (int i = 0; i < 4; i++) {
            #pragma unroll
            for (int r = 0; r < 8; r++) {
                int row = bm + mw * 64 + i * 16 + r + rofs;
                float v = acc[i][j][r] + bb;
                if (mode == 0) {
                    outH[(size_t)row * N + col] = (h16)v;
                } else if (mode == 1) {
                    float g = 0.5f * v * (1.f + erff(v * 0.70710678118654752f));
                    outH[(size_t)row * N + col] = (h16)g;
                } else if (mode == 2) {
                    outF[(size_t)row * N + col] = res[(size_t)row * N + col] + v;
                } else {
                    int tk = src_token(row);
                    outF[(size_t)tk * N + col] = res[(size_t)tk * N + col] + v;
                }
            }
        }
    }
}

// ------------------------- windowed attention ------------------------------
__device__ __forceinline__ int regidx(int h) { return (h < 21) ? 0 : ((h < 25) ? 1 : 2); }

__global__ __launch_bounds__(128) void attn_kernel(
    const h16* __restrict__ qkv,     // [1024*49, 1536]
    const float* __restrict__ rpb,   // [169, 16]
    h16* __restrict__ out) {         // [1024*49, 512]
    int head = blockIdx.x;
    int win  = blockIdx.y;
    __shared__ __align__(16) h16  qs[64 * 32];
    __shared__ __align__(16) h16  ks[64 * 32];
    __shared__ __align__(16) h16  vT[32 * 64];
    __shared__ __align__(16) float sc[64 * 64];
    __shared__ __align__(16) h16  p[64 * 64];

    int tid = threadIdx.x, lane = tid & 31, wv = tid >> 5;
    size_t rowbase = (size_t)win * NTOK;
    const float scale = 0.17677669529663687f;  // 32^-0.5

    for (int idx = tid; idx < 64 * 32; idx += 128) {
        int r = idx >> 5, d = idx & 31;
        if (r < NTOK) {
            const h16* base = qkv + (rowbase + r) * 1536 + head * HD + d;
            qs[idx]        = (h16)((float)base[0] * scale);
            ks[idx]        = base[512];
            vT[d * 64 + r] = base[1024];
        } else {
            qs[idx] = (h16)0.f; ks[idx] = (h16)0.f; vT[d * 64 + r] = (h16)0.f;
        }
    }
    __syncthreads();

    int lrow = lane & 15;
    int rofs = (lane >= 16) ? 8 : 0;
    int wim = win & 15, wh = wim >> 2, ww = wim & 3;
    v16h aq = frag_a(&qs[(wv * 16 + lrow) * 32], lane);

    #pragma unroll
    for (int j = 0; j < 4; j++) {
        v16h bk = frag_b(&ks[(j * 16 + lrow) * 32], lane);
        v8f a = {};
        a = __builtin_amdgcn_wmma_f32_16x16x32_f16(false, aq, false, bk,
                                                   (short)0, a, false, false);
        int col = j * 16 + lrow;
        #pragma unroll
        for (int r = 0; r < 8; r++) {
            int row = wv * 16 + r + rofs;
            float v = a[r];
            if (row < NTOK && col < NTOK) {
                int i1 = row / 7, j1 = row % 7, i2 = col / 7, j2 = col % 7;
                int rel = (i1 - i2 + 6) * 13 + (j1 - j2 + 6);
                v += rpb[rel * NHEAD + head];
                int c1 = regidx(wh * 7 + i1) * 3 + regidx(ww * 7 + j1);
                int c2 = regidx(wh * 7 + i2) * 3 + regidx(ww * 7 + j2);
                if (c1 != c2) v -= 100.f;
            } else {
                v = -1e30f;
            }
            sc[row * 64 + col] = v;
        }
    }
    __syncthreads();

    if (tid < 64) {
        float mx = -1e30f;
        for (int c = 0; c < NTOK; c++) mx = fmaxf(mx, sc[tid * 64 + c]);
        float sum = 0.f;
        for (int c = 0; c < NTOK; c++) sum += __expf(sc[tid * 64 + c] - mx);
        float inv = 1.f / sum;
        for (int c = 0; c < NTOK; c++)
            p[tid * 64 + c] = (h16)(__expf(sc[tid * 64 + c] - mx) * inv);
        for (int c = NTOK; c < 64; c++) p[tid * 64 + c] = (h16)0.f;
    }
    __syncthreads();

    v8f o[2] = {};
    #pragma unroll
    for (int kstep = 0; kstep < 2; kstep++) {
        v16h ap = frag_a(&p[(wv * 16 + lrow) * 64 + kstep * 32], lane);
        #pragma unroll
        for (int j = 0; j < 2; j++) {
            v16h bv = frag_b(&vT[(j * 16 + lrow) * 64 + kstep * 32], lane);
            o[j] = __builtin_amdgcn_wmma_f32_16x16x32_f16(false, ap, false, bv,
                                                          (short)0, o[j], false, false);
        }
    }
    #pragma unroll
    for (int j = 0; j < 2; j++) {
        int col = j * 16 + lrow;  // d in [0,32)
        #pragma unroll
        for (int r = 0; r < 8; r++) {
            int row = wv * 16 + r + rofs;
            if (row < NTOK)
                out[(rowbase + row) * C_DIM + head * HD + col] = (h16)o[j][r];
        }
    }
}

// ------------------------------- launcher ----------------------------------
extern "C" void kernel_launch(void* const* d_in, const int* in_sizes, int n_in,
                              void* d_out, int out_size, void* d_ws, size_t ws_size,
                              hipStream_t stream) {
    const float* x      = (const float*)d_in[0];
    const float* ln1_s  = (const float*)d_in[1];
    const float* ln1_b  = (const float*)d_in[2];
    const float* qkv_w  = (const float*)d_in[3];
    const float* qkv_b  = (const float*)d_in[4];
    const float* qkv_la = (const float*)d_in[5];
    const float* qkv_lb = (const float*)d_in[6];
    const float* rpb    = (const float*)d_in[7];
    const float* proj_w = (const float*)d_in[8];
    const float* proj_b = (const float*)d_in[9];
    const float* proj_la= (const float*)d_in[10];
    const float* proj_lb= (const float*)d_in[11];
    const float* ln2_s  = (const float*)d_in[12];
    const float* ln2_b  = (const float*)d_in[13];
    const float* fc1_w  = (const float*)d_in[14];
    const float* fc1_b  = (const float*)d_in[15];
    const float* fc1_la = (const float*)d_in[16];
    const float* fc1_lb = (const float*)d_in[17];
    const float* fc2_w  = (const float*)d_in[18];
    const float* fc2_b  = (const float*)d_in[19];
    const float* fc2_la = (const float*)d_in[20];
    const float* fc2_lb = (const float*)d_in[21];

    char* ws = (char*)d_ws;
    size_t off = 0;
    auto alloc = [&](size_t bytes) {
        void* pp = ws + off;
        off += (bytes + 255) & ~(size_t)255;
        return pp;
    };
    h16* wqkv   = (h16*)alloc((size_t)1536 * 512 * 2);
    h16* wproj  = (h16*)alloc((size_t)512 * 512 * 2);
    h16* wfc1   = (h16*)alloc((size_t)2048 * 512 * 2);
    h16* wfc2   = (h16*)alloc((size_t)512 * 2048 * 2);
    h16* xw     = (h16*)alloc((size_t)MROWS * 512 * 2);     // also reused as xn2
    h16* qkvbuf = (h16*)alloc((size_t)MROWS * 1536 * 2);    // [qkvbuf|attnout] reused as h
    h16* attnout= (h16*)alloc((size_t)MROWS * 512 * 2);
    float* x1   = (float*)alloc((size_t)MROWS * 512 * 4);
    h16* xn2    = xw;          // alias (xw dead after QKV GEMM)
    h16* hbuf   = qkvbuf;      // alias, 50176*2048*2 == qkvbuf+attnout extents

    // 1) fold LoRA into weights (f16, [N][K])
    fuse_w_kernel<<<(1536 * 512 + 255) / 256, 256, 0, stream>>>(qkv_w, qkv_la, qkv_lb, wqkv, 1536, 512);
    fuse_w_kernel<<<(512 * 512 + 255) / 256, 256, 0, stream>>>(proj_w, proj_la, proj_lb, wproj, 512, 512);
    fuse_w_kernel<<<(2048 * 512 + 255) / 256, 256, 0, stream>>>(fc1_w, fc1_la, fc1_lb, wfc1, 2048, 512);
    fuse_w_kernel<<<(512 * 2048 + 255) / 256, 256, 0, stream>>>(fc2_w, fc2_la, fc2_lb, wfc2, 512, 2048);

    // 2) LN1 + cyclic shift + window partition -> xw (f16, window order)
    ln_kernel<<<MROWS, 256, 0, stream>>>(x, ln1_s, ln1_b, xw, 1);

    // 3) QKV GEMM: [50176,512] x [512,1536]
    gemm_wmma_kernel<<<dim3(MROWS / BM, 1536 / BN), 256, 0, stream>>>(
        xw, wqkv, qkv_b, MROWS, 1536, 512, 0, qkvbuf, nullptr, nullptr);

    // 4) attention per (head, window)
    attn_kernel<<<dim3(NHEAD, 1024), 128, 0, stream>>>(qkvbuf, rpb, attnout);

    // 5) proj GEMM + window-reverse + reverse-shift + residual -> x1 (fp32)
    gemm_wmma_kernel<<<dim3(MROWS / BM, 512 / BN), 256, 0, stream>>>(
        attnout, wproj, proj_b, MROWS, 512, 512, 3, nullptr, x1, x);

    // 6) LN2 -> xn2 (f16)
    ln_kernel<<<MROWS, 256, 0, stream>>>(x1, ln2_s, ln2_b, xn2, 0);

    // 7) fc1 GEMM + exact GELU -> h (f16)
    gemm_wmma_kernel<<<dim3(MROWS / BM, 2048 / BN), 256, 0, stream>>>(
        xn2, wfc1, fc1_b, MROWS, 2048, 512, 1, hbuf, nullptr, nullptr);

    // 8) fc2 GEMM + residual -> d_out (fp32)
    gemm_wmma_kernel<<<dim3(MROWS / BM, 512 / BN), 256, 0, stream>>>(
        hbuf, wfc2, fc2_b, MROWS, 512, 2048, 2, nullptr, (float*)d_out, x1);
}
